// LSTMEncoder_25881472925893
// MI455X (gfx1250) — matrix operand order repaired and checked by reading, compiled, not run on hardware
//
#include <hip/hip_runtime.h>
#include <hip/hip_bf16.h>

// MI455X / gfx1250 LSTM encoder.
// ws layout: seq16 (64MB) | hring (64KB) | barriers (256B) | x16 (64MB, optional)

typedef __attribute__((ext_vector_type(16))) _Float16 v16h;
typedef __attribute__((ext_vector_type(8)))  _Float16 v8h;
typedef __attribute__((ext_vector_type(2)))  _Float16 h2;
typedef __attribute__((ext_vector_type(8)))  float    v8f;

#define BATCH 64
#define TLEN  2048
#define HID   256
#define INP   256
#define OUTD  256
#define NGATE 1024
#define DEPTH 2
#define NWG   8
#define OUTSZ ((size_t)BATCH * TLEN * OUTD)

#define SEQ_BYTES   ((size_t)TLEN * BATCH * HID * 2)       // 64 MB
#define HRING_OFF   SEQ_BYTES
#define HRING_BYTES ((size_t)2 * BATCH * HID * 2)          // 64 KB
#define BAR_OFF     (HRING_OFF + HRING_BYTES)
#define X16_OFF     (BAR_OFF + 256)
#define X16_BYTES   ((size_t)BATCH * TLEN * INP * 2)       // 64 MB
#define WS_NEED_BIG (X16_OFF + X16_BYTES)

#if defined(__HIP_DEVICE_COMPILE__) && __has_builtin(__builtin_amdgcn_global_load_async_to_lds_b128)
#define HAVE_ASYNC 1
#else
#define HAVE_ASYNC 0
#endif

// pointer types matching the async-to-LDS builtin signature:
// (int4 addrspace(1)*, int4 addrspace(3)*, Imm offset, Imm cpol)
typedef __attribute__((__vector_size__(16))) int i4_t;
typedef __attribute__((address_space(1))) i4_t i4_glb;
typedef __attribute__((address_space(3))) i4_t i4_lds;

// per-lane 16B global -> LDS direct copy (ASYNCcnt-tracked) with VALU fallback
__device__ __forceinline__ void async_cp16(_Float16* lds, const _Float16* g) {
#if HAVE_ASYNC
  __builtin_amdgcn_global_load_async_to_lds_b128((i4_glb*)g, (i4_lds*)lds, 0, 0);
#else
  *reinterpret_cast<v8h*>(lds) = *reinterpret_cast<const v8h*>(g);
#endif
}
__device__ __forceinline__ void wait_async0() {
#if HAVE_ASYNC
#if __has_builtin(__builtin_amdgcn_s_wait_asynccnt)
  __builtin_amdgcn_s_wait_asynccnt(0);
#else
  asm volatile("s_wait_asynccnt 0x0" ::: "memory");
#endif
#endif
}

__device__ __forceinline__ float sigf(float x) {
  return 1.0f / (1.0f + __expf(-x));
}
__device__ __forceinline__ float tanh_(float x) {
  x = fminf(fmaxf(x, -15.0f), 15.0f);
  float e = __expf(2.0f * x);
  return (e - 1.0f) / (e + 1.0f);
}

// A-fragment gather (16x32 f16 A layout, ISA 7.12.2): two aligned ds_load_b128
template <int LDA>
__device__ __forceinline__ v16h load_a_frag(const _Float16* As, int mt, int kc, int lane) {
  const int m  = mt * 16 + (lane & 15);
  const int hi = (lane >> 4) & 1;
  const _Float16* row = As + m * LDA;
  v8h lo = *reinterpret_cast<const v8h*>(row + kc * 32 + hi * 8);
  v8h hh = *reinterpret_cast<const v8h*>(row + kc * 32 + 16 + hi * 8);
  return __builtin_shufflevector(lo, hh, 0,1,2,3,4,5,6,7,8,9,10,11,12,13,14,15);
}

// device-wide barrier among NWG persistent WGs; relaxed polls, fences at edges
__device__ __forceinline__ void gbar(unsigned* cnt, unsigned target) {
  __syncthreads();
  if (threadIdx.x == 0) {
    __builtin_amdgcn_fence(__ATOMIC_RELEASE, "agent");
    __hip_atomic_fetch_add(cnt, 1u, __ATOMIC_RELAXED, __HIP_MEMORY_SCOPE_AGENT);
    while (__hip_atomic_load(cnt, __ATOMIC_RELAXED, __HIP_MEMORY_SCOPE_AGENT) < target) {
      __builtin_amdgcn_s_sleep(2);
    }
    __builtin_amdgcn_fence(__ATOMIC_ACQUIRE, "agent");
  }
  __syncthreads();
}

__global__ void lstm_init_k(unsigned* bar) {
  if (threadIdx.x < 16) bar[threadIdx.x] = 0u;
}

// x [B,T,INP] f32 -> x16 f16 (same layout); 8 elems/thread
__global__ __launch_bounds__(256) void cvt_x_k(const float* __restrict__ x,
                                               _Float16* __restrict__ x16) {
  const size_t i = ((size_t)blockIdx.x * 256 + threadIdx.x) * 8;
  float4 a = *reinterpret_cast<const float4*>(x + i);
  float4 b = *reinterpret_cast<const float4*>(x + i + 4);
  v8h o;
  o[0] = (_Float16)a.x; o[1] = (_Float16)a.y; o[2] = (_Float16)a.z; o[3] = (_Float16)a.w;
  o[4] = (_Float16)b.x; o[5] = (_Float16)b.y; o[6] = (_Float16)b.z; o[7] = (_Float16)b.w;
  *reinterpret_cast<v8h*>(x16 + i) = o;
}

// One LSTM layer; grid = NWG x 256. WG g owns h-cols [g*32, g*32+32).
// Wave -> one 16-col gate N-tile, 4 M-tiles, K=512, weights VGPR-resident.
// One device barrier per step; h exchanged via hring ping-pong; layer1 defers
// its seq[t] overwrite by one step (held in registers).
__global__ __launch_bounds__(256, 1) void lstm_layer_k(
    const float* __restrict__ xin, const _Float16* __restrict__ x16,
    const float* __restrict__ w_ih, const float* __restrict__ w_hh,
    const float* __restrict__ b_ih, const float* __restrict__ b_hh,
    _Float16* __restrict__ seq16, _Float16* __restrict__ hring,
    float* __restrict__ dout, unsigned* __restrict__ bar,
    int layer, int use_x16) {
  constexpr int LDA = 520;
  __shared__ _Float16 As[BATCH][LDA];
  __shared__ float gS[BATCH][128];
  __shared__ float cS[BATCH][32];
  __shared__ float biasS[128];

  const int tid  = threadIdx.x;
  const int lane = tid & 31;
  const int w    = tid >> 5;
  const int g    = blockIdx.x;

  if (tid < 128) {
    int qq = tid >> 5, jj = tid & 31;
    int n = qq * 256 + g * 32 + jj;
    biasS[tid] = b_ih[layer * NGATE + n] + b_hh[layer * NGATE + n];
  }
  for (int i = tid; i < BATCH * 32; i += 256) (&cS[0][0])[i] = 0.0f;

  // W slice -> register B-fragments (32x16 f16 B layout)
  const int q = w >> 1, sub = w & 1;
  const int ncol = q * 256 + g * 32 + sub * 16 + (lane & 15);
  const float* Wih = w_ih + (size_t)layer * NGATE * INP + (size_t)ncol * INP;
  const float* Whh = w_hh + (size_t)layer * NGATE * HID + (size_t)ncol * HID;
  v16h bw[16];
#pragma unroll
  for (int kc = 0; kc < 16; ++kc) {
    const int kb = kc * 32 + ((lane >> 4) & 1) * 8;
    v16h f;
#pragma unroll
    for (int e = 0; e < 16; ++e) {
      const int k = kb + e + (e >= 8 ? 8 : 0);
      f[e] = (_Float16)((k < INP) ? Wih[k] : Whh[k - INP]);
    }
    bw[kc] = f;
  }
  __syncthreads();

  float hold[8];
#pragma unroll
  for (int i = 0; i < 8; ++i) hold[i] = 0.0f;

  for (int t = 0; t < TLEN; ++t) {
    const size_t seq_t  = (size_t)t * BATCH * HID;
    const size_t seq_tm = (size_t)(t - 1) * BATCH * HID;
    _Float16* hcur        = hring + (size_t)(t & 1) * BATCH * HID;
    const _Float16* hprev = hring + (size_t)((t & 1) ^ 1) * BATCH * HID;

    // ---- stage A = [x_t | h_{t-1}] (f16) ----
    if (layer == 1 || use_x16) {
      for (int p = tid; p < 2048; p += 256) {
        const int b = p >> 5, kk = (p & 31) * 8;
        const _Float16* gp = (layer == 0)
            ? x16 + (size_t)b * TLEN * INP + (size_t)t * INP + kk
            : seq16 + seq_t + (size_t)b * HID + kk;
        async_cp16(&As[b][kk], gp);
      }
    } else {   // layer 0, f32 convert fallback (small ws)
      for (int p = tid; p < 8192; p += 256) {
        const int b = p >> 7, k = (p & 127) * 2;
        float2 xv = *reinterpret_cast<const float2*>(
            xin + (size_t)b * TLEN * INP + (size_t)t * INP + k);
        h2 v; v[0] = (_Float16)xv.x; v[1] = (_Float16)xv.y;
        *reinterpret_cast<h2*>(&As[b][k]) = v;
      }
    }
    if (t == 0) {
      v8h z;
#pragma unroll
      for (int e = 0; e < 8; ++e) z[e] = (_Float16)0.0f;
      for (int p = tid; p < 2048; p += 256) {
        const int b = p >> 5, kk = (p & 31) * 8;
        *reinterpret_cast<v8h*>(&As[b][256 + kk]) = z;
      }
    } else {
      for (int p = tid; p < 2048; p += 256) {
        const int b = p >> 5, kk = (p & 31) * 8;
        async_cp16(&As[b][256 + kk], hprev + (size_t)b * HID + kk);
      }
    }
    wait_async0();
    __syncthreads();

    // ---- GEMM over K=512: v_wmma_f32_16x16x32_f16 ----
    v8f zf = {0.f, 0.f, 0.f, 0.f, 0.f, 0.f, 0.f, 0.f};
    v8f acc[4];
#pragma unroll
    for (int i = 0; i < 4; ++i) acc[i] = zf;
#pragma unroll
    for (int kc = 0; kc < 16; ++kc) {
      v16h a0 = load_a_frag<LDA>(&As[0][0], 0, kc, lane);
      v16h a1 = load_a_frag<LDA>(&As[0][0], 1, kc, lane);
      v16h a2 = load_a_frag<LDA>(&As[0][0], 2, kc, lane);
      v16h a3 = load_a_frag<LDA>(&As[0][0], 3, kc, lane);
      acc[0] = __builtin_amdgcn_wmma_f32_16x16x32_f16(false, a0, false, bw[kc], (short)0, acc[0], false, false);
      acc[1] = __builtin_amdgcn_wmma_f32_16x16x32_f16(false, a1, false, bw[kc], (short)0, acc[1], false, false);
      acc[2] = __builtin_amdgcn_wmma_f32_16x16x32_f16(false, a2, false, bw[kc], (short)0, acc[2], false, false);
      acc[3] = __builtin_amdgcn_wmma_f32_16x16x32_f16(false, a3, false, bw[kc], (short)0, acc[3], false, false);
    }

    // ---- spill gate tiles to LDS (C/D f32 layout) ----
#pragma unroll
    for (int mt = 0; mt < 4; ++mt)
#pragma unroll
      for (int r = 0; r < 8; ++r) {
        const int m = mt * 16 + r + ((lane >> 4) & 1) * 8;
        gS[m][w * 16 + (lane & 15)] = acc[mt][r];
      }
    __syncthreads();

    // ---- gate combine + cell update ----
    int ii = 0;
    for (int i = tid; i < BATCH * 32; i += 256, ++ii) {
      const int m = i >> 5, jj = i & 31;
      const int gcol = g * 32 + jj;
      const float iv = sigf (gS[m][jj]      + biasS[jj]);
      const float fv = sigf (gS[m][32 + jj] + biasS[32 + jj]);
      const float gv = tanh_(gS[m][64 + jj] + biasS[64 + jj]);
      const float ov = sigf (gS[m][96 + jj] + biasS[96 + jj]);
      const float c  = fv * cS[m][jj] + iv * gv;
      cS[m][jj] = c;
      const float h = ov * tanh_(c);
      hcur[(size_t)m * HID + gcol] = (_Float16)h;
      if (layer == 0) {
        seq16[seq_t + (size_t)m * HID + gcol] = (_Float16)h;
      } else {
        if (t > 0) seq16[seq_tm + (size_t)m * HID + gcol] = (_Float16)hold[ii];
        hold[ii] = h;
      }
      if (t == TLEN - 1) {
        const size_t base = OUTSZ + (size_t)layer * BATCH * HID + (size_t)m * HID + gcol;
        dout[base] = h;                                   // h_n
        dout[base + (size_t)DEPTH * BATCH * HID] = c;     // c_n
      }
    }
    gbar(bar, (unsigned)(t + 1) * NWG);   // single barrier per step
  }

  if (layer == 1) {   // flush deferred final h into seq[T-1]
    int ii = 0;
    for (int i = tid; i < BATCH * 32; i += 256, ++ii) {
      const int m = i >> 5, jj = i & 31;
      seq16[(size_t)(TLEN - 1) * BATCH * HID + (size_t)m * HID + g * 32 + jj] =
          (_Float16)hold[ii];
    }
  }
}

// out[b,t,o] = PReLU(seq1[t,b,:] . w_fc[o,:] + b_fc[o])
__global__ __launch_bounds__(256, 1) void fc_prelu_k(
    const _Float16* __restrict__ seq16,
    const float* __restrict__ w_fc, const float* __restrict__ b_fc,
    const float* __restrict__ prelu_a, float* __restrict__ out) {
  constexpr int LDA = 264;
  __shared__ _Float16 As[BATCH][LDA];

  const int tid  = threadIdx.x;
  const int lane = tid & 31;
  const int w    = tid >> 5;
  const int tT   = blockIdx.x >> 1;
  const int half = blockIdx.x & 1;
  const size_t base = (size_t)tT * BATCH * HID;

  for (int p = tid; p < 2048; p += 256) {
    const int b = p >> 5, kk = (p & 31) * 8;
    async_cp16(&As[b][kk], seq16 + base + (size_t)b * HID + kk);
  }

  const int n = half * 128 + w * 16 + (lane & 15);
  const float* Wf = w_fc + (size_t)n * HID;
  v16h bw[8];
#pragma unroll
  for (int kc = 0; kc < 8; ++kc) {
    const int kb = kc * 32 + ((lane >> 4) & 1) * 8;
    v16h f;
#pragma unroll
    for (int e = 0; e < 16; ++e) f[e] = (_Float16)Wf[kb + e + (e >= 8 ? 8 : 0)];
    bw[kc] = f;
  }
  wait_async0();
  __syncthreads();

  v8f zf = {0.f, 0.f, 0.f, 0.f, 0.f, 0.f, 0.f, 0.f};
  v8f acc[4];
#pragma unroll
  for (int i = 0; i < 4; ++i) acc[i] = zf;
#pragma unroll
  for (int kc = 0; kc < 8; ++kc) {
    v16h a0 = load_a_frag<LDA>(&As[0][0], 0, kc, lane);
    v16h a1 = load_a_frag<LDA>(&As[0][0], 1, kc, lane);
    v16h a2 = load_a_frag<LDA>(&As[0][0], 2, kc, lane);
    v16h a3 = load_a_frag<LDA>(&As[0][0], 3, kc, lane);
    acc[0] = __builtin_amdgcn_wmma_f32_16x16x32_f16(false, a0, false, bw[kc], (short)0, acc[0], false, false);
    acc[1] = __builtin_amdgcn_wmma_f32_16x16x32_f16(false, a1, false, bw[kc], (short)0, acc[1], false, false);
    acc[2] = __builtin_amdgcn_wmma_f32_16x16x32_f16(false, a2, false, bw[kc], (short)0, acc[2], false, false);
    acc[3] = __builtin_amdgcn_wmma_f32_16x16x32_f16(false, a3, false, bw[kc], (short)0, acc[3], false, false);
  }

  const float bfv = b_fc[n];
  const float pa  = prelu_a[0];
#pragma unroll
  for (int mt = 0; mt < 4; ++mt)
#pragma unroll
    for (int r = 0; r < 8; ++r) {
      const int b = mt * 16 + r + ((lane >> 4) & 1) * 8;
      float v = acc[mt][r] + bfv;
      v = (v >= 0.0f) ? v : pa * v;
      out[(size_t)b * TLEN * OUTD + (size_t)tT * OUTD + n] = v;
    }
}

extern "C" void kernel_launch(void* const* d_in, const int* in_sizes, int n_in,
                              void* d_out, int out_size, void* d_ws, size_t ws_size,
                              hipStream_t stream) {
  (void)in_sizes; (void)n_in; (void)out_size;
  const float* x    = (const float*)d_in[0];
  const float* w_ih = (const float*)d_in[1];
  const float* w_hh = (const float*)d_in[2];
  const float* b_ih = (const float*)d_in[3];
  const float* b_hh = (const float*)d_in[4];
  const float* w_fc = (const float*)d_in[5];
  const float* b_fc = (const float*)d_in[6];
  const float* pra  = (const float*)d_in[7];
  float* out = (float*)d_out;

  _Float16* seq16 = (_Float16*)d_ws;
  _Float16* hring = (_Float16*)((char*)d_ws + HRING_OFF);
  unsigned* bar   = (unsigned*)((char*)d_ws + BAR_OFF);
  _Float16* x16   = (_Float16*)((char*)d_ws + X16_OFF);
  const int use_x16 = (ws_size >= WS_NEED_BIG) ? 1 : 0;

  lstm_init_k<<<1, 64, 0, stream>>>(bar);
  if (use_x16) {
    cvt_x_k<<<(int)((size_t)BATCH * TLEN * INP / (256 * 8)), 256, 0, stream>>>(x, x16);
  }
  lstm_layer_k<<<NWG, 256, 0, stream>>>(x, x16, w_ih, w_hh, b_ih, b_hh,
                                        seq16, hring, out, bar + 0, 0, use_x16);
  lstm_layer_k<<<NWG, 256, 0, stream>>>(x, x16, w_ih, w_hh, b_ih, b_hh,
                                        seq16, hring, out, bar + 1, 1, use_x16);
  fc_prelu_k<<<TLEN * 2, 256, 0, stream>>>(seq16, w_fc, b_fc, pra, out);
}
